// DKAModule_27066883899461
// MI455X (gfx1250) — matrix operand order repaired and checked by reading, not compiled, not run on hardware
//
#include <hip/hip_runtime.h>
#include <hip/hip_bf16.h>
#include <math.h>

// ---------------------------------------------------------------------------
// Problem constants (from reference)
// ---------------------------------------------------------------------------
#define BSZ      2
#define SEQ      4096
#define DMODEL   1024
#define NHEAD    8
#define DHEAD    128
#define RANK     4
#define KMAXC    21
#define MROWS    (BSZ * SEQ)          // 8192

typedef __attribute__((ext_vector_type(16))) _Float16 v16h;
typedef __attribute__((ext_vector_type(8)))  _Float16 half8;
typedef __attribute__((ext_vector_type(4)))  _Float16 half4;
typedef __attribute__((ext_vector_type(8)))  float    v8f;
typedef __attribute__((ext_vector_type(4)))  int      v4i;

typedef __attribute__((address_space(1))) v4i as1_v4i;  // global int4
typedef __attribute__((address_space(3))) v4i as3_v4i;  // LDS int4

#if __has_builtin(__builtin_amdgcn_global_load_async_to_lds_b128) && \
    __has_builtin(__builtin_amdgcn_s_wait_asynccnt)
#define HAS_ASYNC_LDS 1
#else
#define HAS_ASYNC_LDS 0
#endif

// ---------------------------------------------------------------------------
// f32 -> f16 conversion (vectorized, 4 elems/thread)
// ---------------------------------------------------------------------------
__global__ void cvt_f32_to_f16(const float* __restrict__ src,
                               _Float16* __restrict__ dst, int n4) {
    int i = blockIdx.x * blockDim.x + threadIdx.x;
    if (i < n4) {
        float4 v = ((const float4*)src)[i];
        half4 o;
        o.x = (_Float16)v.x; o.y = (_Float16)v.y;
        o.z = (_Float16)v.z; o.w = (_Float16)v.w;
        ((half4*)dst)[i] = o;
    }
}

// ---------------------------------------------------------------------------
// WMMA GEMM:  C[M,N] = A[M,K](f16) * B[N,K]^T(f16) + bias[N]
// Block = 256 threads = 8 waves.  BM=128, BN=128, BK=32.
// Wave grid 2(M) x 4(N); each wave -> 64x32 patch = 4x2 WMMA 16x16 tiles.
// LDS row stride 40 halves (80B, 16B-aligned, bank-spread).
// Double-buffered async Global->LDS DMA (ASYNCcnt) when available.
// ---------------------------------------------------------------------------
#define BK 32
#define LDSST 40
#define TILEH (128 * LDSST)           // halves per tile buffer

__device__ __forceinline__ void async_cp_b128(const _Float16* g, _Float16* l) {
#if HAS_ASYNC_LDS
    __builtin_amdgcn_global_load_async_to_lds_b128(
        (as1_v4i*)(_Float16*)g,       // drop const, cast to AS1 int4*
        (as3_v4i*)l, 0, 0);
#else
    (void)g; (void)l;
#endif
}

__global__ __launch_bounds__(256)
void gemm_f16_wmma(const _Float16* __restrict__ A,
                   const _Float16* __restrict__ Bw,
                   const float*    __restrict__ bias,
                   float*          __restrict__ C,
                   int M, int N, int K) {
    // two ping-pong buffers, A and B tiles each
    __shared__ _Float16 sA[2 * TILEH];
    __shared__ _Float16 sB[2 * TILEH];

    const int tid   = threadIdx.x;
    const int wave  = tid >> 5;
    const int lane  = tid & 31;
    const int waveM = wave >> 2;        // 0..1 -> 64 rows
    const int waveN = wave & 3;         // 0..3 -> 32 cols
    const int blockM = blockIdx.y * 128;
    const int blockN = blockIdx.x * 128;

    // cooperative tile-load mapping: 256 threads x 32 halves
    const int ldRow = tid >> 1;         // 0..127
    const int ldCol = (tid & 1) * 16;   // 0 or 16

    v8f acc[4][2];
#pragma unroll
    for (int i = 0; i < 4; ++i)
#pragma unroll
        for (int j = 0; j < 2; ++j) acc[i][j] = {};

    const _Float16* ga = A  + (size_t)(blockM + ldRow) * K + ldCol;
    const _Float16* gb = Bw + (size_t)(blockN + ldRow) * K + ldCol;
    const int lOff = ldRow * LDSST + ldCol;

    const int mr   = lane & 15;
    const int ksel = lane >> 4;

#if HAS_ASYNC_LDS
    // ---- async double-buffered pipeline ----
    // prologue: DMA tile 0 into buffer 0 (4 async ops per wave)
    async_cp_b128(ga,     &sA[lOff]);
    async_cp_b128(ga + 8, &sA[lOff + 8]);
    async_cp_b128(gb,     &sB[lOff]);
    async_cp_b128(gb + 8, &sB[lOff + 8]);

    for (int kt = 0; kt < K; kt += BK) {
        const int cur = (kt >> 5) & 1;
        const int nxt = kt + BK;
        if (nxt < K) {
            const int nb = (nxt >> 5) & 1;
            async_cp_b128(ga + nxt,     &sA[nb * TILEH + lOff]);
            async_cp_b128(ga + nxt + 8, &sA[nb * TILEH + lOff + 8]);
            async_cp_b128(gb + nxt,     &sB[nb * TILEH + lOff]);
            async_cp_b128(gb + nxt + 8, &sB[nb * TILEH + lOff + 8]);
            if (nxt + BK < K) {         // pre-warm L2 one tile further ahead
                __builtin_prefetch(ga + nxt + BK, 0, 1);
                __builtin_prefetch(gb + nxt + BK, 0, 1);
            }
            __builtin_amdgcn_s_wait_asynccnt(4);   // tile-t DMAs retired
        } else {
            __builtin_amdgcn_s_wait_asynccnt(0);
        }
        __syncthreads();

        const _Float16* cA = &sA[cur * TILEH];
        const _Float16* cB = &sB[cur * TILEH];

        // A fragments: 16x32, lane(m = lane&15): K = ksel*8 + e (e<8),
        //                                        K = ksel*8 + 16 + (e-8)
        v16h af[4];
#pragma unroll
        for (int i = 0; i < 4; ++i) {
            int off = (waveM * 64 + i * 16 + mr) * LDSST + ksel * 8;
            half8 lo = *(const half8*)&cA[off];
            half8 hi = *(const half8*)&cA[off + 16];
            af[i] = __builtin_shufflevector(lo, hi, 0,1,2,3,4,5,6,7,
                                                     8,9,10,11,12,13,14,15);
        }
        // B fragments: 32x16, lane(n = lane&15): K = ksel*8 + e (e<8), ksel*8 + 16 + (e-8)
        v16h bf[2];
#pragma unroll
        for (int j = 0; j < 2; ++j) {
            int off = (waveN * 32 + j * 16 + mr) * LDSST + ksel * 8;
            half8 lo = *(const half8*)&cB[off];
            half8 hi = *(const half8*)&cB[off + 16];
            bf[j] = __builtin_shufflevector(lo, hi, 0,1,2,3,4,5,6,7,
                                                     8,9,10,11,12,13,14,15);
        }
#pragma unroll
        for (int i = 0; i < 4; ++i)
#pragma unroll
            for (int j = 0; j < 2; ++j)
                acc[i][j] = __builtin_amdgcn_wmma_f32_16x16x32_f16(
                    false, af[i], false, bf[j], (short)0, acc[i][j],
                    false, false);
        __syncthreads();   // before DMA overwrites the other buffer next iter
    }
#else
    // ---- synchronous fallback (single buffer) ----
    _Float16* la = &sA[lOff];
    _Float16* lb = &sB[lOff];
    for (int kt = 0; kt < K; kt += BK) {
        *(half8*)(la)     = *(const half8*)(ga + kt);
        *(half8*)(la + 8) = *(const half8*)(ga + kt + 8);
        *(half8*)(lb)     = *(const half8*)(gb + kt);
        *(half8*)(lb + 8) = *(const half8*)(gb + kt + 8);
        if (kt + BK < K) {
            __builtin_prefetch(ga + kt + BK, 0, 1);
            __builtin_prefetch(gb + kt + BK, 0, 1);
        }
        __syncthreads();
        v16h af[4];
#pragma unroll
        for (int i = 0; i < 4; ++i) {
            int off = (waveM * 64 + i * 16 + mr) * LDSST + ksel * 8;
            half8 lo = *(const half8*)&sA[off];
            half8 hi = *(const half8*)&sA[off + 16];
            af[i] = __builtin_shufflevector(lo, hi, 0,1,2,3,4,5,6,7,
                                                     8,9,10,11,12,13,14,15);
        }
        v16h bf[2];
#pragma unroll
        for (int j = 0; j < 2; ++j) {
            int off = (waveN * 32 + j * 16 + mr) * LDSST + ksel * 8;
            half8 lo = *(const half8*)&sB[off];
            half8 hi = *(const half8*)&sB[off + 16];
            bf[j] = __builtin_shufflevector(lo, hi, 0,1,2,3,4,5,6,7,
                                                     8,9,10,11,12,13,14,15);
        }
#pragma unroll
        for (int i = 0; i < 4; ++i)
#pragma unroll
            for (int j = 0; j < 2; ++j)
                acc[i][j] = __builtin_amdgcn_wmma_f32_16x16x32_f16(
                    false, af[i], false, bf[j], (short)0, acc[i][j],
                    false, false);
        __syncthreads();
    }
#endif

    // epilogue: VGPR v -> M = v + (lane>>4)*8, N = lane&15
    const int rHi = (lane >> 4) * 8;
    const int nc  = lane & 15;
#pragma unroll
    for (int j = 0; j < 2; ++j) {
        int col = blockN + waveN * 32 + j * 16 + nc;
        float bv = bias[col];
#pragma unroll
        for (int i = 0; i < 4; ++i) {
            int rbase = blockM + waveM * 64 + i * 16 + rHi;
#pragma unroll
            for (int v = 0; v < 8; ++v)
                C[(size_t)(rbase + v) * N + col] = acc[i][j][v] + bv;
        }
    }
}

// ---------------------------------------------------------------------------
// Dynamic-kernel conv, collapsed form:
//   o[d] = a * sum_r c_r * V[r,d] * (sum_k A[r,k]*w[k,d])
//        + (1-a) * sum_k base[k,d]*w[k,d]
// One block per (b,n); 256 threads -> 1024 (h,d) outputs (4 each).
// ---------------------------------------------------------------------------
__global__ __launch_bounds__(256)
void dka_conv(const float* __restrict__ xproj,
              const float* __restrict__ Wc,      // [H][DHEAD][RANK]
              const float* __restrict__ Amat,    // [H][RANK][KMAXC]
              const float* __restrict__ Vmat,    // [H][RANK][DHEAD]
              const float* __restrict__ basem,   // [H][KMAXC][DHEAD]
              const float* __restrict__ alphas,  // [H]
              _Float16* __restrict__ concat16) {
    __shared__ float sx[DMODEL];
    __shared__ float sc[NHEAD][RANK];

    const int p = blockIdx.x;
    const int b = p / SEQ;
    const int n = p - b * SEQ;
    const float* xrow = xproj + (size_t)p * DMODEL;

    for (int i = threadIdx.x; i < DMODEL; i += 256) sx[i] = xrow[i];
    __syncthreads();

    if (threadIdx.x < NHEAD * RANK) {
        int h = threadIdx.x >> 2, r = threadIdx.x & 3;
        const float* wc = Wc + h * (DHEAD * RANK) + r;
        const float* xh = &sx[h * DHEAD];
        float s = 0.f;
#pragma unroll 8
        for (int d = 0; d < DHEAD; ++d) s += xh[d] * wc[d * RANK];
        sc[h][r] = s;
    }
    __syncthreads();

#pragma unroll
    for (int q = 0; q < 4; ++q) {
        const int idx = threadIdx.x + q * 256;       // 0..1023
        const int h = idx >> 7, d = idx & 127;
        const int hp = h >> 1;
        const int kh  = (hp == 0) ? 3 : (hp == 1) ? 7 : (hp == 2) ? 11 : 21;
        const int pad = kh >> 1;

        const float* Ah = Amat  + h * (RANK * KMAXC);
        const float* bh = basem + h * (KMAXC * DHEAD) + d;

        float S0 = 0.f, S1 = 0.f, S2 = 0.f, S3 = 0.f, bacc = 0.f;
        for (int k = 0; k < kh; ++k) {
            int nn = n + k - pad;
            float w = 0.f;
            if (nn >= 0 && nn < SEQ)
                w = xproj[(size_t)(b * SEQ + nn) * DMODEL + h * DHEAD + d];
            bacc += bh[k * DHEAD] * w;
            S0 += Ah[k] * w;
            S1 += Ah[KMAXC + k] * w;
            S2 += Ah[2 * KMAXC + k] * w;
            S3 += Ah[3 * KMAXC + k] * w;
        }
        float alpha = 1.f / (1.f + __expf(-alphas[h]));
        const float* Vh = Vmat + h * (RANK * DHEAD) + d;
        float dyn = sc[h][0] * Vh[0]         * S0
                  + sc[h][1] * Vh[DHEAD]     * S1
                  + sc[h][2] * Vh[2 * DHEAD] * S2
                  + sc[h][3] * Vh[3 * DHEAD] * S3;
        float o = alpha * dyn + (1.f - alpha) * bacc;
        concat16[(size_t)p * DMODEL + idx] = (_Float16)o;
    }
}

// ---------------------------------------------------------------------------
// Host-side launcher
// ---------------------------------------------------------------------------
extern "C" void kernel_launch(void* const* d_in, const int* in_sizes, int n_in,
                              void* d_out, int out_size, void* d_ws, size_t ws_size,
                              hipStream_t stream) {
    const float* x      = (const float*)d_in[0];
    const float* W_in   = (const float*)d_in[1];
    const float* b_in   = (const float*)d_in[2];
    const float* W_out  = (const float*)d_in[3];
    const float* b_out  = (const float*)d_in[4];
    const float* Wc     = (const float*)d_in[5];
    const float* Amat   = (const float*)d_in[6];
    const float* Vmat   = (const float*)d_in[7];
    const float* basem  = (const float*)d_in[8];
    const float* alphas = (const float*)d_in[9];
    float* out = (float*)d_out;

    // workspace layout (bytes)
    char* ws = (char*)d_ws;
    const size_t SZ_X16   = (size_t)MROWS * DMODEL * sizeof(_Float16);   // 16 MB
    const size_t SZ_W16   = (size_t)DMODEL * DMODEL * sizeof(_Float16);  //  2 MB
    const size_t SZ_PROJ  = (size_t)MROWS * DMODEL * sizeof(float);      // 32 MB
    _Float16* xh16     = (_Float16*)(ws);
    _Float16* win16    = (_Float16*)(ws + SZ_X16);
    _Float16* wout16   = (_Float16*)(ws + SZ_X16 + SZ_W16);
    float*    xproj    = (float*)   (ws + SZ_X16 + 2 * SZ_W16);
    _Float16* concat16 = (_Float16*)(ws + SZ_X16 + 2 * SZ_W16 + SZ_PROJ);

    // 1) precision conversion for WMMA operands
    {
        int n4 = MROWS * DMODEL / 4;
        cvt_f32_to_f16<<<(n4 + 255) / 256, 256, 0, stream>>>(x, xh16, n4);
        int w4 = DMODEL * DMODEL / 4;
        cvt_f32_to_f16<<<(w4 + 255) / 256, 256, 0, stream>>>(W_in, win16, w4);
        cvt_f32_to_f16<<<(w4 + 255) / 256, 256, 0, stream>>>(W_out, wout16, w4);
    }

    // 2) x_proj = x @ W_in^T + b_in   (WMMA f16 -> f32)
    {
        dim3 grid(DMODEL / 128, MROWS / 128);
        gemm_f16_wmma<<<grid, 256, 0, stream>>>(xh16, win16, b_in, xproj,
                                                MROWS, DMODEL, DMODEL);
    }

    // 3) dynamic-kernel conv -> concat (f16)
    dka_conv<<<MROWS, 256, 0, stream>>>(xproj, Wc, Amat, Vmat, basem,
                                        alphas, concat16);

    // 4) out = concat @ W_out^T + b_out
    {
        dim3 grid(DMODEL / 128, MROWS / 128);
        gemm_f16_wmma<<<grid, 256, 0, stream>>>(concat16, wout16, b_out, out,
                                                MROWS, DMODEL, DMODEL);
    }
    (void)in_sizes; (void)n_in; (void)out_size; (void)ws_size;
}
